// DownConv_24739011625522
// MI455X (gfx1250) — matrix operand order repaired and checked by reading, not compiled, not run on hardware
//
#include <hip/hip_runtime.h>

typedef __attribute__((ext_vector_type(16))) __bf16 v16bf;
typedef __attribute__((ext_vector_type(8)))  float  v8f;
typedef __attribute__((ext_vector_type(4)))  unsigned u32x4;
typedef __attribute__((ext_vector_type(8)))  int      i32x8;
typedef __attribute__((ext_vector_type(4)))  int      i32x4;

#define DEVI __device__ __forceinline__

constexpr int BB = 2, NP = 100000, R = 256, CIN = 64, COUT = 128;

// d_out layout (floats): out | before_pool | x_after | c
constexpr long long OUT_SZ = (long long)BB * COUT * 128 * 128;     // 4194304
constexpr long long BP_OFF = OUT_SZ;
constexpr long long BP_SZ  = (long long)BB * COUT * R * R;         // 16777216
constexpr long long XA_OFF = BP_OFF + BP_SZ;                       // 20971520
constexpr long long C_OFF  = XA_OFF + BP_SZ;                       // 37748736

// workspace layout (bytes)
constexpr size_t WC1_OFF  = 0;                           // [3][3][128][64] bf16
constexpr size_t WC2_OFF  = WC1_OFF + 9*128*64*2;        // [3][3][128][128] bf16
constexpr size_t W11_OFF  = WC2_OFF + 9*128*128*2;       // [128][64] bf16
constexpr size_t WFC1_OFF = W11_OFF + 128*64*2;          // [256][128] bf16 (T)
constexpr size_t WFC2_OFF = WFC1_OFF + 256*128*2;        // [128][256] bf16 (T)
constexpr size_t WFCC_OFF = WFC2_OFF + 128*256*2;        // [128][64] bf16 (T)
constexpr size_t CNT_OFF  = WFCC_OFF + 128*64*2;         // [2][65536] f32
constexpr size_t H1_OFF   = CNT_OFF + (size_t)BB*65536*4; // [2][256][256][128] bf16

// ---------------- CDNA5 async + TDM helpers ----------------
// Flat pointers into the LDS aperture carry the wave-relative LDS byte offset
// in their low 32 bits (ISA 10.2).
DEVI unsigned lds_off_u32(const void* p) { return (unsigned)(uintptr_t)p; }

DEVI void async_l2g_b128(const void* gaddr, unsigned lds_off) {
  asm volatile("global_store_async_from_lds_b128 %0, %1, off"
               :: "v"((unsigned long long)(uintptr_t)gaddr), "v"(lds_off)
               : "memory");
}
DEVI void wait_async0() { asm volatile("s_wait_asynccnt 0x0" ::: "memory"); }

// Tensor DMA descriptor (ISA cdna5 ch.8): group0 = count/lds/global/type,
// group1 = data_size(1B)/dims/strides, groups 2-3 zero (<=2D tiles).
// This toolchain exposes the 6-arg builtin form (extra int32x8, zero-filled).
DEVI u32x4 tdm_g0(unsigned lds_addr, unsigned long long gaddr) {
  u32x4 g;
  g[0] = 1u;                                             // count=1 (valid)
  g[1] = lds_addr;                                       // bits[63:32]
  g[2] = (unsigned)(gaddr & 0xFFFFFFFFu);                // global_addr[31:0]
  g[3] = (unsigned)((gaddr >> 32) & 0x01FFFFFFu) | 0x80000000u; // [56:32]|type=2
  return g;
}
DEVI i32x8 tdm_g1(unsigned long long tdim0, unsigned tdim1,
                  unsigned tile0, unsigned tile1, unsigned long long stride0) {
  i32x8 g = {};
  g[0] = 0;                                              // mask=0, data_size=1B
  g[1] = (int)((tdim0 & 0xFFFFu) << 16);                 // tensor_dim0[15:0]
  g[2] = (int)(((tdim0 >> 16) & 0xFFFFu) | ((tdim1 & 0xFFFFu) << 16));
  g[3] = (int)(((tdim1 >> 16) & 0xFFFFu) | (tile0 << 16));   // tile_dim0
  g[4] = (int)(tile1 & 0xFFFFu);                         // tile_dim1, tile_dim2=0
  g[5] = (int)(stride0 & 0xFFFFFFFFu);                   // dim0_stride[31:0]
  g[6] = (int)((stride0 >> 32) & 0xFFFFu);               // dim0_stride[47:32]
  g[7] = 0;
  return g;
}
DEVI void tdm_load2d(unsigned lds, const void* g, unsigned long long tdim0,
                     unsigned tdim1, unsigned tile0, unsigned tile1,
                     unsigned long long stride0) {
  i32x4 z4 = {};
  i32x8 z8 = {};
  __builtin_amdgcn_tensor_load_to_lds(
      tdm_g0(lds, (unsigned long long)(uintptr_t)g),
      tdm_g1(tdim0, tdim1, tile0, tile1, stride0), z4, z4, z8, 0);
}
DEVI void tdm_store2d(unsigned lds, const void* g, unsigned long long tdim0,
                      unsigned tdim1, unsigned tile0, unsigned tile1,
                      unsigned long long stride0) {
  i32x4 z4 = {};
  i32x8 z8 = {};
  __builtin_amdgcn_tensor_store_from_lds(
      tdm_g0(lds, (unsigned long long)(uintptr_t)g),
      tdm_g1(tdim0, tdim1, tile0, tile1, stride0), z4, z4, z8, 0);
}
DEVI void wait_tensor0() { __builtin_amdgcn_s_wait_tensorcnt((short)0); }

// ---------------- WMMA fragment helpers (ISA 7.12.2 layouts) ----------------
DEVI v16bf frag_a(const __bf16* A, int strideM, int lane) {
  int m = lane & 15, half = lane >> 4;
  v16bf a;
#pragma unroll
  for (int e = 0; e < 16; ++e) {
    int k = (e < 8) ? (half * 8 + e) : (half * 8 + e + 8);
    a[e] = A[m * strideM + k];
  }
  return a;
}
DEVI v16bf frag_b(const __bf16* BT, int strideN, int lane) {
  int n = lane & 15, kb = (lane >> 4) * 8;
  v16bf b;
#pragma unroll
  for (int e = 0; e < 16; ++e) b[e] = BT[n * strideN + kb + e + ((e < 8) ? 0 : 8)];
  return b;
}
DEVI v8f wmma_bf16(v16bf a, v16bf b, v8f c) {
  return __builtin_amdgcn_wmma_f32_16x16x32_bf16(false, a, false, b, (short)0, c,
                                                 false, false);
}

// ---------------- weight conversion ----------------
__global__ void k_cvt_conv_w(const float* __restrict__ src, __bf16* __restrict__ dst,
                             int ci_n) {
  int total = 128 * ci_n * 9;
  for (int t = blockIdx.x * blockDim.x + threadIdx.x; t < total;
       t += gridDim.x * blockDim.x) {
    int kx = t % 3, ky = (t / 3) % 3, ci = (t / 9) % ci_n, co = t / (9 * ci_n);
    dst[((ky * 3 + kx) * 128 + co) * ci_n + ci] =
        (__bf16)src[((co * ci_n + ci) * 3 + ky) * 3 + kx];
  }
}
__global__ void k_cvt_T(const float* __restrict__ src, __bf16* __restrict__ dst,
                        int K, int N) {
  for (int t = blockIdx.x * blockDim.x + threadIdx.x; t < K * N;
       t += gridDim.x * blockDim.x) {
    int n = t % N, k = t / N;
    dst[n * K + k] = (__bf16)src[t];
  }
}
__global__ void k_cvt_plain(const float* __restrict__ src, __bf16* __restrict__ dst,
                            int n) {
  for (int t = blockIdx.x * blockDim.x + threadIdx.x; t < n;
       t += gridDim.x * blockDim.x)
    dst[t] = (__bf16)src[t];
}

// ---------------- conv1: 3x3 64->128 + relu, implicit GEMM ----------------
__global__ __launch_bounds__(256) void k_conv1(const float* __restrict__ xin,
                                               const __bf16* __restrict__ wc1,
                                               const float* __restrict__ b1,
                                               __bf16* __restrict__ h1) {
  __shared__ alignas(64) __bf16 s_in[3][18][64];
  __shared__ alignas(64) __bf16 s_o1[16][128];  // [px][co], contiguous 4KB tile
  const int b  = blockIdx.y;
  const int y  = blockIdx.x >> 4;
  const int x0 = (blockIdx.x & 15) * 16;
  const int tid = threadIdx.x;
  for (int t = tid; t < 3 * 18 * 64; t += 256) {
    int px = t % 18, row = (t / 18) % 3, ci = t / 54;
    int xx = x0 - 1 + px, yy = y - 1 + row;
    float v = 0.f;
    if (xx >= 0 && xx < R && yy >= 0 && yy < R)
      v = xin[(((long long)b * CIN + ci) * R + yy) * R + xx];
    s_in[row][px][ci] = (__bf16)v;
  }
  __syncthreads();
  const int wave = tid >> 5, lane = tid & 31;
  const int co0 = wave * 16, nn = lane & 15, half = lane >> 4;
  v8f acc = {};
#pragma unroll
  for (int ky = 0; ky < 3; ++ky)
#pragma unroll
    for (int kx = 0; kx < 3; ++kx) {
      const __bf16* bp = wc1 + ((ky * 3 + kx) * 128 + co0) * 64;
#pragma unroll
      for (int kc = 0; kc < 2; ++kc) {
        v16bf a  = frag_a(&s_in[ky][kx][kc * 32], 64, lane);
        v16bf bf = frag_b(bp + kc * 32, 64, lane);
        acc = wmma_bf16(a, bf, acc);
      }
    }
  float bias = b1[co0 + nn];
#pragma unroll
  for (int r = 0; r < 8; ++r) {
    int m = r + half * 8;
    float v = acc[r] + bias;
    v = v > 0.f ? v : 0.f;
    s_o1[m][co0 + nn] = (__bf16)v;
  }
  __syncthreads();
  {  // async LDS->global store: output tile is one contiguous 4KB block
    const char* gbase =
        (const char*)(h1 + (((long long)b * R + y) * R + x0) * COUT);
    async_l2g_b128(gbase + tid * 16, lds_off_u32(&s_o1[0][0]) + tid * 16);
    wait_async0();
  }
}

// ---------------- conv2: 3x3 128->128 (+relu) + 1x1 64->128, fused add ----
__global__ __launch_bounds__(256) void k_conv2(const __bf16* __restrict__ h1,
                                               const float* __restrict__ xpc,
                                               const __bf16* __restrict__ wc2,
                                               const __bf16* __restrict__ w11,
                                               const float* __restrict__ b2,
                                               const float* __restrict__ b11,
                                               float* __restrict__ xafter) {
  __shared__ alignas(64) __bf16 s2[3][18][128];
  __shared__ alignas(64) __bf16 s1[16][64];
  __shared__ alignas(64) float  s_out[128][16];  // [co][px]
  const int b  = blockIdx.y;
  const int y  = blockIdx.x >> 4;
  const int xt = blockIdx.x & 15;
  const int x0 = xt * 16;
  const int tid = threadIdx.x;
  const bool interior = (y >= 1) && (y <= R - 2) && (xt >= 1) && (xt <= 14);
  if (interior) {
    // One TDM 2D-tile load per WG: 3 rows x 4608B, row stride 64KB.
    if (tid < 32) {
      const __bf16* g = h1 + (((long long)b * R + (y - 1)) * R + (x0 - 1)) * COUT;
      tdm_load2d(lds_off_u32(&s2[0][0][0]), g,
                 /*tensor_dim0=*/(unsigned long long)R * COUT * 2,
                 /*tensor_dim1=*/(unsigned)(BB * R),
                 /*tile0=*/18 * COUT * 2, /*tile1=*/3,
                 /*stride0=*/(unsigned long long)R * COUT * 2);
    }
  } else {
    for (int t = tid; t < 3 * 18 * 128; t += 256) {
      int ci = t & 127, px = (t >> 7) % 18, row = (t >> 7) / 18;
      int xx = x0 - 1 + px, yy = y - 1 + row;
      __bf16 v = (__bf16)0.f;
      if (xx >= 0 && xx < R && yy >= 0 && yy < R)
        v = h1[(((long long)b * R + yy) * R + xx) * COUT + ci];
      s2[row][px][ci] = v;
    }
  }
  for (int t = tid; t < 16 * 64; t += 256) {
    int px = t & 15, ci = t >> 4;
    s1[px][ci] = (__bf16)xpc[(((long long)b * CIN + ci) * R + y) * R + x0 + px];
  }
  if (interior && tid < 32) wait_tensor0();
  __syncthreads();
  const int wave = tid >> 5, lane = tid & 31;
  const int co0 = wave * 16, nn = lane & 15, half = lane >> 4;
  v8f acc = {}, accp = {};
#pragma unroll
  for (int ky = 0; ky < 3; ++ky)
#pragma unroll
    for (int kx = 0; kx < 3; ++kx) {
      const __bf16* bp = wc2 + ((ky * 3 + kx) * 128 + co0) * 128;
#pragma unroll
      for (int kc = 0; kc < 4; ++kc) {
        v16bf a  = frag_a(&s2[ky][kx][kc * 32], 128, lane);
        v16bf bf = frag_b(bp + kc * 32, 128, lane);
        acc = wmma_bf16(a, bf, acc);
      }
    }
#pragma unroll
  for (int kc = 0; kc < 2; ++kc) {
    v16bf a  = frag_a(&s1[0][kc * 32], 64, lane);
    v16bf bf = frag_b(w11 + co0 * 64 + kc * 32, 64, lane);
    accp = wmma_bf16(a, bf, accp);
  }
  float bias2 = b2[co0 + nn], biasp = b11[co0 + nn];
#pragma unroll
  for (int r = 0; r < 8; ++r) {
    int m = r + half * 8;
    float v = acc[r] + bias2;
    v = v > 0.f ? v : 0.f;
    v += accp[r] + biasp;
    s_out[co0 + nn][m] = v;
  }
  __syncthreads();
  if (tid < 32) {
    // One TDM 2D-tile store per WG: 128 rows x 64B, row stride 256KB.
    const float* g = &xafter[(((long long)b * COUT) * R + y) * R + x0];
    tdm_store2d(lds_off_u32(&s_out[0][0]), g,
                /*tensor_dim0=*/(unsigned long long)R * R * 4,
                /*tensor_dim1=*/(unsigned)(BB * COUT),
                /*tile0=*/16 * 4, /*tile1=*/COUT,
                /*stride0=*/(unsigned long long)R * R * 4);
    wait_tensor0();
  }
}

// ------- fused: bilinear sample + fc1(relu) + fc2 + c_last*fcc + scatter ----
__global__ __launch_bounds__(256) void k_points(
    const float* __restrict__ pxyz, const float* __restrict__ xafter,
    const float* __restrict__ clast, const __bf16* __restrict__ wfc1T,
    const float* __restrict__ bfc1, const __bf16* __restrict__ wfc2T,
    const float* __restrict__ bfc2, const __bf16* __restrict__ wfccT,
    const float* __restrict__ bfcc, float* __restrict__ cout_,
    float* __restrict__ plane, float* __restrict__ cnt) {
  __shared__ __bf16 s_c0[16][128];
  __shared__ __bf16 s_cl[16][64];
  __shared__ __bf16 s_h[16][256];
  __shared__ int s_idx[16];
  __shared__ int s_bi[16];
  const int tid = threadIdx.x;
  const long long g0 = (long long)blockIdx.x * 16;
  {  // stage A: sampling + c_last staging + scatter index
    int p = tid >> 4, sub = tid & 15;
    long long row = g0 + p;
    int b = (int)(row / NP);
    float fx = pxyz[row * 3 + 0], fy = pxyz[row * 3 + 1];
    float px = fminf(fmaxf(fx * 255.f, 0.f), 255.f);
    float py = fminf(fmaxf(fy * 255.f, 0.f), 255.f);
    int x0i = (int)floorf(px), y0i = (int)floorf(py);
    int x1i = min(x0i + 1, 255), y1i = min(y0i + 1, 255);
    float wx = px - (float)x0i, wy = py - (float)y0i;
    const float* f = xafter + (long long)b * COUT * R * R;
#pragma unroll
    for (int j = 0; j < 8; ++j) {
      int ch = sub * 8 + j;
      const float* fc = f + (long long)ch * R * R;
      float f00 = fc[y0i * R + x0i], f01 = fc[y0i * R + x1i];
      float f10 = fc[y1i * R + x0i], f11 = fc[y1i * R + x1i];
      float v = f00 * (1.f - wx) * (1.f - wy) + f01 * wx * (1.f - wy) +
                f10 * (1.f - wx) * wy + f11 * wx * wy;
      s_c0[p][ch] = (__bf16)v;
    }
#pragma unroll
    for (int j = 0; j < 4; ++j) {
      int ch = sub * 4 + j;
      s_cl[p][ch] = (__bf16)clast[row * 64 + ch];
    }
    if (sub == 0) {
      int ix = (int)(fx * 256.f);
      ix = ix < 0 ? 0 : (ix > 255 ? 255 : ix);
      int iy = (int)(fy * 256.f);
      iy = iy < 0 ? 0 : (iy > 255 ? 255 : iy);
      s_idx[p] = ix + 256 * iy;
      s_bi[p] = b;
      atomicAdd(&cnt[b * 65536 + ix + 256 * iy], 1.f);
    }
  }
  __syncthreads();
  const int wave = tid >> 5, lane = tid & 31;
  const int nn = lane & 15, half = lane >> 4;
#pragma unroll
  for (int s = 0; s < 2; ++s) {
    int n0 = (wave * 2 + s) * 16;
    v8f acc = {};
#pragma unroll
    for (int kc = 0; kc < 4; ++kc) {
      v16bf a  = frag_a(&s_c0[0][kc * 32], 128, lane);
      v16bf bf = frag_b(wfc1T + n0 * 128 + kc * 32, 128, lane);
      acc = wmma_bf16(a, bf, acc);
    }
    float bias = bfc1[n0 + nn];
#pragma unroll
    for (int r = 0; r < 8; ++r) {
      int m = r + half * 8;
      float v = acc[r] + bias;
      v = v > 0.f ? v : 0.f;
      s_h[m][n0 + nn] = (__bf16)v;
    }
  }
  __syncthreads();
  {
    int n0 = wave * 16;
    v8f acc = {};
#pragma unroll
    for (int kc = 0; kc < 8; ++kc) {
      v16bf a  = frag_a(&s_h[0][kc * 32], 256, lane);
      v16bf bf = frag_b(wfc2T + n0 * 256 + kc * 32, 256, lane);
      acc = wmma_bf16(a, bf, acc);
    }
#pragma unroll
    for (int kc = 0; kc < 2; ++kc) {
      v16bf a  = frag_a(&s_cl[0][kc * 32], 64, lane);
      v16bf bf = frag_b(wfccT + n0 * 64 + kc * 32, 64, lane);
      acc = wmma_bf16(a, bf, acc);
    }
    float bias = bfc2[n0 + nn] + bfcc[n0 + nn];
#pragma unroll
    for (int r = 0; r < 8; ++r) {
      int m = r + half * 8;
      float v = acc[r] + bias;
      long long row = g0 + m;
      cout_[row * 128 + n0 + nn] = v;
      atomicAdd(&plane[((long long)s_bi[m] * COUT + n0 + nn) * 65536 + s_idx[m]], v);
    }
  }
}

// ---------------- scatter-mean finalize + maxpool ----------------
__global__ void k_finalize(float* __restrict__ plane, const float* __restrict__ cnt) {
  long long t = (long long)blockIdx.x * blockDim.x + threadIdx.x;
  if (t >= BP_SZ) return;
  int i = (int)(t & 65535);
  int b = (int)(t >> 23);
  plane[t] = plane[t] / fmaxf(cnt[b * 65536 + i], 1.f);
}
__global__ void k_maxpool(const float* __restrict__ plane, float* __restrict__ out_) {
  long long t = (long long)blockIdx.x * blockDim.x + threadIdx.x;
  if (t >= OUT_SZ) return;
  int x = (int)(t & 127), y = (int)((t >> 7) & 127);
  long long cb = t >> 14;
  const float* s = plane + cb * 65536;
  int y2 = y * 2, x2 = x * 2;
  float m = fmaxf(fmaxf(s[y2 * 256 + x2], s[y2 * 256 + x2 + 1]),
                  fmaxf(s[(y2 + 1) * 256 + x2], s[(y2 + 1) * 256 + x2 + 1]));
  out_[t] = m;
}

extern "C" void kernel_launch(void* const* d_in, const int* in_sizes, int n_in,
                              void* d_out, int out_size, void* d_ws, size_t ws_size,
                              hipStream_t stream) {
  (void)in_sizes; (void)n_in; (void)out_size; (void)ws_size;
  const float* p     = (const float*)d_in[0];
  const float* xxy   = (const float*)d_in[1];
  const float* xac   = (const float*)d_in[2];
  const float* clast = (const float*)d_in[3];
  const float* wc1f  = (const float*)d_in[4];
  const float* b1    = (const float*)d_in[5];
  const float* wc2f  = (const float*)d_in[6];
  const float* b2    = (const float*)d_in[7];
  const float* w11f  = (const float*)d_in[8];
  const float* b11   = (const float*)d_in[9];
  const float* wfc1  = (const float*)d_in[10];
  const float* bfc1  = (const float*)d_in[11];
  const float* wfc2  = (const float*)d_in[12];
  const float* bfc2  = (const float*)d_in[13];
  const float* wfcc  = (const float*)d_in[14];
  const float* bfcc  = (const float*)d_in[15];
  float* out = (float*)d_out;
  char*  ws  = (char*)d_ws;
  __bf16* wc1   = (__bf16*)(ws + WC1_OFF);
  __bf16* wc2   = (__bf16*)(ws + WC2_OFF);
  __bf16* w11   = (__bf16*)(ws + W11_OFF);
  __bf16* wfc1T = (__bf16*)(ws + WFC1_OFF);
  __bf16* wfc2T = (__bf16*)(ws + WFC2_OFF);
  __bf16* wfccT = (__bf16*)(ws + WFCC_OFF);
  float*  cnt   = (float*)(ws + CNT_OFF);
  __bf16* h1    = (__bf16*)(ws + H1_OFF);

  k_cvt_conv_w<<<64, 256, 0, stream>>>(wc1f, wc1, 64);
  k_cvt_conv_w<<<64, 256, 0, stream>>>(wc2f, wc2, 128);
  k_cvt_plain<<<32, 256, 0, stream>>>(w11f, w11, 128 * 64);
  k_cvt_T<<<64, 256, 0, stream>>>(wfc1, wfc1T, 128, 256);
  k_cvt_T<<<64, 256, 0, stream>>>(wfc2, wfc2T, 256, 128);
  k_cvt_T<<<16, 256, 0, stream>>>(wfcc, wfccT, 64, 128);

  (void)hipMemsetAsync(out + BP_OFF, 0, (size_t)BP_SZ * sizeof(float), stream);
  (void)hipMemsetAsync(cnt, 0, (size_t)BB * 65536 * sizeof(float), stream);

  k_conv1<<<dim3(4096, BB), 256, 0, stream>>>(xxy, wc1, b1, h1);
  k_conv2<<<dim3(4096, BB), 256, 0, stream>>>(h1, xac, wc2, w11, b2, b11,
                                              out + XA_OFF);
  k_points<<<(BB * NP) / 16, 256, 0, stream>>>(p, out + XA_OFF, clast, wfc1T, bfc1,
                                               wfc2T, bfc2, wfccT, bfcc,
                                               out + C_OFF, out + BP_OFF, cnt);
  k_finalize<<<(int)(BP_SZ / 256), 256, 0, stream>>>(out + BP_OFF, cnt);
  k_maxpool<<<(int)(OUT_SZ / 256), 256, 0, stream>>>(out + BP_OFF, out);
}
